// precalcSDE_66228395704879
// MI455X (gfx1250) — compile-verified, matrix operand order checked by reading, not run on hardware
//
#include <hip/hip_runtime.h>
#include <hip/hip_bf16.h>

#define NUM_WORDS   31995
#define W_PAD       32000
#define DIM         128
#define LATENT      10000
#define LPAD        10016      // LATENT padded to chunk boundary (zeros)
#define TOTAL_NG    639900
#define PAD_IDX     1

typedef __attribute__((ext_vector_type(16))) __bf16 v16bf;
typedef __attribute__((ext_vector_type(8)))  __bf16 v8bf;
typedef __attribute__((ext_vector_type(8)))  float  v8f;

union V16 { v16bf v; v8bf h[2]; };

// ---------------------------------------------------------------------------
// Kernel A: EmbeddingBag(mean) + tanh.  segment_ids sorted -> binary search.
// One block (128 threads, one per dim) per word; pads words >= NUM_WORDS to 0.
// ---------------------------------------------------------------------------
__global__ void bag_mean_tanh(const int* __restrict__ ngram_ids,
                              const int* __restrict__ seg,
                              const float* __restrict__ table,
                              float* __restrict__ lang_f32,
                              __bf16* __restrict__ lang_bf) {
  const int w = blockIdx.x;
  const int d = threadIdx.x;
  float val = 0.f;
  if (w < NUM_WORDS) {
    int lo = 0, hi = TOTAL_NG;
    while (lo < hi) { int m = (lo + hi) >> 1; if (seg[m] < w) lo = m + 1; else hi = m; }
    int lo2 = lo, hi2 = TOTAL_NG;
    while (lo2 < hi2) { int m = (lo2 + hi2) >> 1; if (seg[m] < w + 1) lo2 = m + 1; else hi2 = m; }
    const int cnt = lo2 - lo;
    float s = 0.f;
    for (int i = lo; i < lo2; ++i) {
      const int id = ngram_ids[i];
      s += table[id * DIM + d];
    }
    val = tanhf(cnt > 0 ? s / (float)cnt : 0.f);
  }
  lang_f32[w * DIM + d] = val;
  lang_bf [w * DIM + d] = (__bf16)val;
}

// ---------------------------------------------------------------------------
// Kernel B: convert codebook to bf16, both row-major [LPAD,DIM] (for Q.K^T B
// operand) and transposed [DIM,LPAD] (for P.V B operand). Pad rows are zero.
// ---------------------------------------------------------------------------
__global__ void cvt_latent(const float* __restrict__ latent,
                           __bf16* __restrict__ lat_bf,
                           __bf16* __restrict__ latT_bf) {
  const int idx = blockIdx.x * blockDim.x + threadIdx.x;
  if (idx >= LPAD * DIM) return;
  const int l = idx / DIM, d = idx % DIM;
  const float v = (l < LATENT) ? latent[idx] : 0.f;
  const __bf16 b = (__bf16)v;
  lat_bf[idx] = b;
  latT_bf[d * LPAD + l] = b;
}

// ---------------------------------------------------------------------------
// Kernel S: special rows (0..3) and mask row (31999) of the embedding table.
// ---------------------------------------------------------------------------
__global__ void write_specials(const float* __restrict__ spec,
                               const float* __restrict__ maskv,
                               float* __restrict__ emb) {
  const int t = blockIdx.x * blockDim.x + threadIdx.x;
  if (t < 4 * DIM)        emb[t] = spec[t];
  else if (t < 5 * DIM)   emb[(W_PAD - 1) * DIM + (t - 4 * DIM)] = maskv[t - 4 * DIM];
}

// ---------------------------------------------------------------------------
// Kernel C: fused streaming softmax-attention over the codebook.
// One wave32 per 32-word (2 x 16 M-tile) block: every B fragment (K-tile for
// Q.K^T, V-tile for P.V) is amortized over TWO independent WMMAs (32 FLOP/B
// from L2). B fragments are double-buffered so the next load is in flight
// while the current WMMA pair executes. bf16 WMMA (K=32), f32 accumulation.
// No running max needed: |score| is tiny, exp is safe; denominators deferred
// to a final butterfly reduction. P tiles transposed to A-operand layout via
// a 2KB LDS staging buffer (compiler tracks the DScnt dependency).
// ---------------------------------------------------------------------------
__global__ void __launch_bounds__(32) latent_attn(
    const __bf16* __restrict__ lang_bf,     // [W_PAD][DIM]
    const float*  __restrict__ lang_f32,    // [W_PAD][DIM]
    const __bf16* __restrict__ lat_bf,      // [LPAD][DIM]
    const __bf16* __restrict__ latT_bf,     // [DIM][LPAD]
    float* __restrict__ emb) {              // [W_PAD][DIM]
  __shared__ __attribute__((aligned(16))) __bf16 Pbuf[2][16 * 32];

  const int lane = threadIdx.x;
  const int n    = lane & 15;   // column within 16-lane half
  const int hi   = lane >> 4;   // half-wave selector (K-group / row-group)
  const int w0   = blockIdx.x * 32;

  // A fragments: 2 M-tiles x 128 dims, loop-invariant (2 x 4 x v16bf)
  V16 afrag[2][4];
#pragma unroll
  for (int m = 0; m < 2; ++m) {
    const __bf16* row = lang_bf + (w0 + m * 16 + n) * DIM;
#pragma unroll
    for (int kk = 0; kk < 4; ++kk) {
      afrag[m][kk].h[0] = *(const v8bf*)(row + kk * 32 + 8 * hi);
      afrag[m][kk].h[1] = *(const v8bf*)(row + kk * 32 + 16 + 8 * hi);
    }
  }

  v8f o[2][8];
  {
    v8f z = {};
#pragma unroll
    for (int m = 0; m < 2; ++m)
#pragma unroll
      for (int i = 0; i < 8; ++i) o[m][i] = z;
  }
  float psum[2][8];
#pragma unroll
  for (int m = 0; m < 2; ++m)
#pragma unroll
    for (int j = 0; j < 8; ++j) psum[m][j] = 0.f;

  const int NCHUNK = LATENT / 32;                 // 312 full chunks + 16 tail
  for (int ch = 0; ch <= NCHUNK; ++ch) {
    const int l0 = ch * 32;
    const int ntiles = (ch < NCHUNK) ? 2 : 1;

    // pull next chunk's K/V tiles toward the caches while this chunk computes
    if (ch < NCHUNK) {
      __builtin_prefetch(lat_bf + (l0 + 32 + n) * DIM, 0, 0);
      __builtin_prefetch(lat_bf + (l0 + 48 + n) * DIM, 0, 0);
      __builtin_prefetch(latT_bf + (8 * n) * LPAD + l0 + 32, 0, 0);
    }

    for (int t = 0; t < ntiles; ++t) {
      // Q.K^T: shared B operand = latent rows (l0 + t*16 + n); double-buffered
      v8f s0 = {}, s1 = {};
      const __bf16* brow = lat_bf + (l0 + t * 16 + n) * DIM;
      V16 bf[2];
      bf[0].h[0] = *(const v8bf*)(brow + 8 * hi);
      bf[0].h[1] = *(const v8bf*)(brow + 16 + 8 * hi);
#pragma unroll
      for (int kk = 0; kk < 4; ++kk) {
        if (kk < 3) {
          bf[(kk + 1) & 1].h[0] = *(const v8bf*)(brow + (kk + 1) * 32 + 8 * hi);
          bf[(kk + 1) & 1].h[1] = *(const v8bf*)(brow + (kk + 1) * 32 + 16 + 8 * hi);
        }
        s0 = __builtin_amdgcn_wmma_f32_16x16x32_bf16(
                 false, afrag[0][kk].v, false, bf[kk & 1].v, (short)0, s0, false, false);
        s1 = __builtin_amdgcn_wmma_f32_16x16x32_bf16(
                 false, afrag[1][kk].v, false, bf[kk & 1].v, (short)0, s1, false, false);
      }
      // exp (no max subtraction), per-lane denom partials, stash P tiles
#pragma unroll
      for (int j = 0; j < 8; ++j) {
        const float p0 = __expf(s0[j]);
        const float p1 = __expf(s1[j]);
        psum[0][j] += p0;
        psum[1][j] += p1;
        Pbuf[0][(j + 8 * hi) * 32 + t * 16 + n] = (__bf16)p0;
        Pbuf[1][(j + 8 * hi) * 32 + t * 16 + n] = (__bf16)p1;
      }
    }
    if (ntiles == 1) {  // tail: zero K columns 16..31 of both P tiles
      v8bf z = {};
      *(v8bf*)(&Pbuf[0][n * 32 + 16 + 8 * hi]) = z;
      *(v8bf*)(&Pbuf[1][n * 32 + 16 + 8 * hi]) = z;
    }

    // reload P tiles as A operands (16x32 bf16); compiler inserts DScnt waits
    V16 ap0, ap1;
    ap0.h[0] = *(const v8bf*)(&Pbuf[0][n * 32 + 8 * hi]);
    ap0.h[1] = *(const v8bf*)(&Pbuf[0][n * 32 + 16 + 8 * hi]);
    ap1.h[0] = *(const v8bf*)(&Pbuf[1][n * 32 + 8 * hi]);
    ap1.h[1] = *(const v8bf*)(&Pbuf[1][n * 32 + 16 + 8 * hi]);

    // P.V: shared B operand from transposed codebook; double-buffered
    const __bf16* vbase = latT_bf + n * LPAD + l0;
    V16 bv[2];
    bv[0].h[0] = *(const v8bf*)(vbase + 8 * hi);
    bv[0].h[1] = *(const v8bf*)(vbase + 16 + 8 * hi);
#pragma unroll
    for (int dt = 0; dt < 8; ++dt) {
      if (dt < 7) {
        const __bf16* vrow = vbase + (dt + 1) * 16 * LPAD;
        bv[(dt + 1) & 1].h[0] = *(const v8bf*)(vrow + 8 * hi);
        bv[(dt + 1) & 1].h[1] = *(const v8bf*)(vrow + 16 + 8 * hi);
      }
      o[0][dt] = __builtin_amdgcn_wmma_f32_16x16x32_bf16(
                     false, ap0.v, false, bv[dt & 1].v, (short)0, o[0][dt], false, false);
      o[1][dt] = __builtin_amdgcn_wmma_f32_16x16x32_bf16(
                     false, ap1.v, false, bv[dt & 1].v, (short)0, o[1][dt], false, false);
    }
  }

  // row denominators: butterfly within each 16-lane half
#pragma unroll
  for (int m = 0; m < 2; ++m)
#pragma unroll
    for (int j = 0; j < 8; ++j) {
      float v = psum[m][j];
      v += __shfl_xor(v, 1, 32);
      v += __shfl_xor(v, 2, 32);
      v += __shfl_xor(v, 4, 32);
      v += __shfl_xor(v, 8, 32);
      psum[m][j] = 1.0f / v;
    }

  // token_emb = lang_emb + latent_emb, written into emb rows 4..31998
#pragma unroll
  for (int m = 0; m < 2; ++m)
#pragma unroll
    for (int dt = 0; dt < 8; ++dt) {
      const int d = dt * 16 + n;
#pragma unroll
      for (int j = 0; j < 8; ++j) {
        const int w = w0 + m * 16 + j + 8 * hi;
        if (w < NUM_WORDS) {
          emb[(4 + w) * DIM + d] = lang_f32[w * DIM + d] + o[m][dt][j] * psum[m][j];
        }
      }
    }
}

// ---------------------------------------------------------------------------
// Kernel E: final gather emb_weight[x]  (padding_idx only affects grad)
// ---------------------------------------------------------------------------
__global__ void gather_out(const int* __restrict__ x,
                           const float* __restrict__ emb,
                           float* __restrict__ out, int total) {
  const int i = blockIdx.x * blockDim.x + threadIdx.x;
  if (i >= total) return;
  const int tok = i / DIM, d = i % DIM;
  int id = x[tok];
  if (id < 0) id = 0; if (id >= W_PAD) id = W_PAD - 1;
  out[i] = emb[id * DIM + d];
}

// ---------------------------------------------------------------------------
extern "C" void kernel_launch(void* const* d_in, const int* in_sizes, int n_in,
                              void* d_out, int out_size, void* d_ws, size_t ws_size,
                              hipStream_t stream) {
  const int*   x           = (const int*)  d_in[0];
  const int*   ngram_ids   = (const int*)  d_in[1];
  const int*   seg         = (const int*)  d_in[2];
  const float* ngram_table = (const float*)d_in[3];
  const float* latent      = (const float*)d_in[4];
  const float* spec        = (const float*)d_in[5];
  const float* maskv       = (const float*)d_in[6];

  // workspace carve-up (all sizes multiples of 256B); total ~44 MB
  char* ws = (char*)d_ws;
  float*  lang_f32 = (float*) (ws);                               // 32000*128*4
  __bf16* lang_bf  = (__bf16*)(ws + 16384000);                    // 32000*128*2
  __bf16* lat_bf   = (__bf16*)(ws + 16384000 + 8192000);          // 10016*128*2
  __bf16* latT_bf  = (__bf16*)(ws + 16384000 + 8192000 + 2564096);// 128*10016*2
  float*  emb      = (float*) (ws + 16384000 + 8192000 + 2564096 + 2564096);

  bag_mean_tanh<<<W_PAD, DIM, 0, stream>>>(ngram_ids, seg, ngram_table,
                                           lang_f32, lang_bf);
  cvt_latent<<<(LPAD * DIM + 255) / 256, 256, 0, stream>>>(latent, lat_bf, latT_bf);
  write_specials<<<(5 * DIM + 127) / 128, 128, 0, stream>>>(spec, maskv, emb);
  latent_attn<<<W_PAD / 32, 32, 0, stream>>>(lang_bf, lang_f32, lat_bf, latT_bf, emb);

  const int total = in_sizes[0] * DIM;   // 4096 tokens * 128
  gather_out<<<(total + 255) / 256, 256, 0, stream>>>(x, emb, (float*)d_out, total);
}
